// DRAWModel_50560355008890
// MI455X (gfx1250) — matrix operand order repaired and checked by reading, not compiled, not run on hardware
//
#include <hip/hip_runtime.h>
#include <hip/hip_bf16.h>
#include <math.h>

typedef __bf16 bf16_t;
typedef __attribute__((ext_vector_type(8)))  bf16_t v8bf;
typedef __attribute__((ext_vector_type(16))) bf16_t v16bf;
typedef __attribute__((ext_vector_type(8)))  float  v8f;

#define T_STEPS 16
#define NBATCH  1024
#define ENCD    512
#define DECD    512
#define ZDIM    128
#define KRAW    637      // READ_FEAT(125) + DEC(512)
#define KENC    640      // padded to multiple of 64
#define NGATE   2048     // 4*512
#define IMGF    4096     // 16^3

#define KSTEP   64       // K-depth per LDS tile (2 WMMA K-chunks)
#define PADK    72       // padded LDS row stride (elements) -> 144B, breaks bank phase

// ---------------------------------------------------------------------------
// CDNA5 async Global->LDS staging + wait helpers
// ---------------------------------------------------------------------------
__device__ inline void async_b128(const bf16_t* g, const bf16_t* lds_generic) {
  // global_load_async_to_lds_b128 vdst(lds byte offset), vaddr(64b global), off
  // Flat LDS aperture keeps the wave-relative LDS address in addr[31:0].
  unsigned int loff = (unsigned int)(unsigned long long)lds_generic;
  asm volatile("global_load_async_to_lds_b128 %0, %1, off"
               :: "v"(loff), "v"(g)
               : "memory");
}

template <int N>
__device__ inline void wait_async() {
  asm volatile("s_wait_asynccnt %0" :: "i"(N) : "memory");
}

// ---------------------------------------------------------------------------
// WMMA fragment loads from LDS tiles (gfx1250 wave32, 16x16x32 bf16 -> f32)
// ---------------------------------------------------------------------------
__device__ inline v16bf lds_frag_a(const bf16_t* tile, int r0, int ks, int lane) {
  // A fragment 16x32: lane&15 = row; K halves split across lane>>4
  int r = r0 + (lane & 15);
  const bf16_t* p = tile + r * PADK + ks + ((lane >> 4) << 3);
  v8bf lo = *(const v8bf*)(p);
  v8bf hi = *(const v8bf*)(p + 16);
  return __builtin_shufflevector(lo, hi, 0,1,2,3,4,5,6,7,8,9,10,11,12,13,14,15);
}

__device__ inline v16bf lds_frag_b(const bf16_t* tile, int r0, int ks, int lane) {
  // B = W^T, W row-major (N x K): lane&15 = output row; K halves split across lane>>4
  int r = r0 + (lane & 15);
  const bf16_t* p = tile + r * PADK + ks + ((lane >> 4) << 3);
  v8bf lo = *(const v8bf*)(p);
  v8bf hi = *(const v8bf*)(p + 16);
  return __builtin_shufflevector(lo, hi, 0,1,2,3,4,5,6,7,8,9,10,11,12,13,14,15);
}

__device__ inline v8f wmma_bf16(v16bf a, v16bf b, v8f c) {
  return __builtin_amdgcn_wmma_f32_16x16x32_bf16(
      /*neg_a=*/false, a, /*neg_b=*/false, b,
      /*c_mod=*/(short)0, c, /*reuse_a=*/false, /*reuse_b=*/false);
}

// ---------------------------------------------------------------------------
// G[M,N] = A1[M,K1] @ W1[N,K1]^T (+ A2[M,K2] @ W2[N,K2]^T) (+ bias0 + bias1)
// block = 256 thr (8 waves); block tile 64(M) x 128(N); wave tile 32x32.
// Double-buffered async LDS staging, K-step 64 (2 WMMA chunks per tile).
// K1, K2 must be multiples of 64.
// ---------------------------------------------------------------------------
__global__ __launch_bounds__(256)
void wmma_gemm_kernel(const bf16_t* __restrict__ A1, const bf16_t* __restrict__ W1, int K1,
                      const bf16_t* __restrict__ A2, const bf16_t* __restrict__ W2, int K2,
                      const float* __restrict__ bias0, const float* __restrict__ bias1,
                      float* __restrict__ G, int N) {
  __shared__ alignas(16) bf16_t As[2][64  * PADK];   // 2 x 9 KB
  __shared__ alignas(16) bf16_t Bs[2][128 * PADK];   // 2 x 18 KB

  const int tid  = threadIdx.x;
  const int lane = tid & 31;
  const int wave = tid >> 5;
  const int wm = wave & 1;        // 0..1 -> 32-row slices
  const int wn = wave >> 1;       // 0..3 -> 32-col slices
  const int nTilesN = N >> 7;
  const int bm = blockIdx.x / nTilesN;
  const int bn = blockIdx.x % nTilesN;
  const int Mbase = bm * 64;
  const int Nbase = bn * 128;
  const int m0 = wm * 32;         // LDS-tile-local
  const int n0 = wn * 32;
  const int nlo = lane & 15;
  const int hi  = lane >> 4;

  const int nt1 = K1 >> 6;
  const int nt2 = A2 ? (K2 >> 6) : 0;
  const int nt  = nt1 + nt2;

  // issue the 6 per-thread async b128 transfers for K-tile i into buffer `buf`
  auto issue = [&](int i, int buf) {
    const bf16_t* Ap; const bf16_t* Wp; int K; int k;
    if (i < nt1) { Ap = A1; Wp = W1; K = K1; k = i * KSTEP; }
    else         { Ap = A2; Wp = W2; K = K2; k = (i - nt1) * KSTEP; }
    // A tile: 64 rows x 64 elems = 512 16B-chunks (8 per row)
#pragma unroll
    for (int c = tid; c < 512; c += 256) {
      int row = c >> 3, off = (c & 7) * 8;
      async_b128(Ap + (size_t)(Mbase + row) * K + k + off,
                 &As[buf][row * PADK + off]);
    }
    // B tile: 128 rows x 64 elems = 1024 16B-chunks
#pragma unroll
    for (int c = tid; c < 1024; c += 256) {
      int row = c >> 3, off = (c & 7) * 8;
      async_b128(Wp + (size_t)(Nbase + row) * K + k + off,
                 &Bs[buf][row * PADK + off]);
    }
  };

  float bv0 = 0.f, bv1 = 0.f;
  if (bias0) { bv0 += bias0[Nbase + n0 + nlo]; bv1 += bias0[Nbase + n0 + 16 + nlo]; }
  if (bias1) { bv0 += bias1[Nbase + n0 + nlo]; bv1 += bias1[Nbase + n0 + 16 + nlo]; }
  v8f c00, c01, c10, c11;
#pragma unroll
  for (int r = 0; r < 8; ++r) { c00[r] = bv0; c01[r] = bv1; c10[r] = bv0; c11[r] = bv1; }

  issue(0, 0);
  int buf = 0;
  for (int i = 0; i < nt; ++i) {
    if (i + 1 < nt) {
      issue(i + 1, buf ^ 1);   // DMA next tile while current is consumed
      wait_async<6>();         // own 6 transfers of tile i complete
    } else {
      wait_async<0>();
    }
    __syncthreads();           // everyone's tile i is in LDS
#pragma unroll
    for (int ks = 0; ks < KSTEP; ks += 32) {
      v16bf a0 = lds_frag_a(As[buf], m0,      ks, lane);
      v16bf a1 = lds_frag_a(As[buf], m0 + 16, ks, lane);
      v16bf b0 = lds_frag_b(Bs[buf], n0,      ks, lane);
      v16bf b1 = lds_frag_b(Bs[buf], n0 + 16, ks, lane);
      c00 = wmma_bf16(a0, b0, c00);
      c01 = wmma_bf16(a0, b1, c01);
      c10 = wmma_bf16(a1, b0, c10);
      c11 = wmma_bf16(a1, b1, c11);
    }
    __syncthreads();           // done with buf before tile i+2 overwrites it
    buf ^= 1;
  }

  // C layout: element r -> M = Mbase+m0 + r + 8*hi, N = Nbase+n0 + (lane&15)
  float* g0 = G + (size_t)(Mbase + m0 + 8 * hi) * N + Nbase + n0 + nlo;
#pragma unroll
  for (int r = 0; r < 8; ++r) {
    g0[(size_t)r * N]             = c00[r];
    g0[(size_t)r * N + 16]        = c01[r];
    g0[(size_t)(r + 16) * N]      = c10[r];
    g0[(size_t)(r + 16) * N + 16] = c11[r];
  }
}

// ---------------------------------------------------------------------------
// Elementwise / sampling kernels
// ---------------------------------------------------------------------------
__device__ inline float sigm(float x) { return 1.f / (1.f + expf(-x)); }

__global__ __launch_bounds__(256)
void lstm_pointwise_kernel(const float* __restrict__ G, float* __restrict__ h,
                           float* __restrict__ c, bf16_t* __restrict__ hb, int H) {
  int idx = blockIdx.x * blockDim.x + threadIdx.x;   // B*H threads
  int b = idx / H, j = idx - b * H;
  const float* g = G + (size_t)b * 4 * H;
  float gi = g[j], gf = g[H + j], gg = g[2 * H + j], go = g[3 * H + j];
  float cn = sigm(gf) * c[idx] + sigm(gi) * tanhf(gg);
  float hn = sigm(go) * tanhf(cn);
  c[idx] = cn;
  h[idx] = hn;
  hb[idx] = (bf16_t)hn;
}

__global__ __launch_bounds__(256)
void z_kernel(const float* __restrict__ MS, const float* __restrict__ mu_b,
              const float* __restrict__ sig_b, const float* __restrict__ e_t,
              bf16_t* __restrict__ zb) {
  int idx = blockIdx.x * blockDim.x + threadIdx.x;   // B*ZDIM
  int b = idx >> 7, j = idx & 127;
  float mu = MS[(size_t)b * 256 + j]        + mu_b[j];
  float ls = MS[(size_t)b * 256 + 128 + j]  + sig_b[j];
  zb[idx] = (bf16_t)(mu + e_t[idx] * expf(ls));
}

__device__ float sample3d(const float* vol, int D, int H, int W,
                          float gx, float gy, float gz) {
  float xf = ((gx + 1.f) * W - 1.f) * 0.5f;
  float yf = ((gy + 1.f) * H - 1.f) * 0.5f;
  float zf = ((gz + 1.f) * D - 1.f) * 0.5f;
  float x0 = floorf(xf), y0 = floorf(yf), z0 = floorf(zf);
  float wx = xf - x0, wy = yf - y0, wz = zf - z0;
  float acc = 0.f;
#pragma unroll
  for (int dz = 0; dz < 2; ++dz)
#pragma unroll
    for (int dy = 0; dy < 2; ++dy)
#pragma unroll
      for (int dx = 0; dx < 2; ++dx) {
        int zi = (int)z0 + dz, yi = (int)y0 + dy, xi = (int)x0 + dx;
        float w = (dz ? wz : 1.f - wz) * (dy ? wy : 1.f - wy) * (dx ? wx : 1.f - wx);
        if (zi >= 0 && zi < D && yi >= 0 && yi < H && xi >= 0 && xi < W)
          acc += vol[(zi * H + yi) * W + xi] * w;
      }
  return acc;
}

// one block per batch element; builds Xcat = bf16([read(h_dec), h_dec, 0,0,0])
__global__ __launch_bounds__(128)
void read_kernel(const float* __restrict__ x, const float* __restrict__ hdec,
                 const float* __restrict__ read_W, const float* __restrict__ read_b,
                 bf16_t* __restrict__ Xcat) {
  int b = blockIdx.x, t = threadIdx.x;
  __shared__ float red[4 * 128];
  __shared__ float P[4];
  const float* h = hdec + (size_t)b * DECD;
  float p0 = 0, p1 = 0, p2 = 0, p3 = 0;
  for (int j = t; j < DECD; j += 128) {
    float hv = h[j];
    p0 += hv * read_W[j];
    p1 += hv * read_W[DECD + j];
    p2 += hv * read_W[2 * DECD + j];
    p3 += hv * read_W[3 * DECD + j];
  }
  red[t] = p0; red[128 + t] = p1; red[256 + t] = p2; red[384 + t] = p3;
  __syncthreads();
  for (int s = 64; s > 0; s >>= 1) {
    if (t < s) {
      red[t] += red[t + s];
      red[128 + t] += red[128 + t + s];
      red[256 + t] += red[256 + t + s];
      red[384 + t] += red[384 + t + s];
    }
    __syncthreads();
  }
  if (t < 4) P[t] = red[t * 128] + read_b[t];
  __syncthreads();
  float s = P[0], tx = P[1], ty = P[2], tz = P[3];
  bf16_t* out = Xcat + (size_t)b * KENC;
  const float* vol = x + (size_t)b * IMGF;
  if (t < 125) {
    int ix = t % 5, iy = (t / 5) % 5, iz = t / 25;
    float gx = ((2.f * ix + 1.f) / 5.f - 1.f) * s + tx;
    float gy = ((2.f * iy + 1.f) / 5.f - 1.f) * s + ty;
    float gz = ((2.f * iz + 1.f) / 5.f - 1.f) * s + tz;
    out[t] = (bf16_t)sample3d(vol, 16, 16, 16, gx, gy, gz);
  }
  for (int j = t; j < DECD; j += 128) out[125 + j] = (bf16_t)h[j];
  if (t < KENC - KRAW) out[KRAW + t] = (bf16_t)0.f;
}

// one block per batch element; canvas += write(h_dec)
__global__ __launch_bounds__(256)
void write_kernel(const float* __restrict__ patchG, const float* __restrict__ w2_b,
                  const float* __restrict__ hdec, const float* __restrict__ w1_W,
                  const float* __restrict__ w1_b, float* __restrict__ canvas) {
  int b = blockIdx.x, t = threadIdx.x;
  __shared__ float red[4 * 256];
  __shared__ float P[4];
  __shared__ float patch[125];
  const float* h = hdec + (size_t)b * DECD;
  float p0 = 0, p1 = 0, p2 = 0, p3 = 0;
  for (int j = t; j < DECD; j += 256) {
    float hv = h[j];
    p0 += hv * w1_W[j];
    p1 += hv * w1_W[DECD + j];
    p2 += hv * w1_W[2 * DECD + j];
    p3 += hv * w1_W[3 * DECD + j];
  }
  red[t] = p0; red[256 + t] = p1; red[512 + t] = p2; red[768 + t] = p3;
  __syncthreads();
  for (int s = 128; s > 0; s >>= 1) {
    if (t < s) {
      red[t] += red[t + s];
      red[256 + t] += red[256 + t + s];
      red[512 + t] += red[512 + t + s];
      red[768 + t] += red[768 + t + s];
    }
    __syncthreads();
  }
  if (t < 4) P[t] = red[t * 256] + w1_b[t];
  if (t < 125) patch[t] = patchG[(size_t)b * 128 + t] + w2_b[t];
  __syncthreads();
  float inv = 1.f / (P[0] + 1e-9f);
  float s = inv;
  float tx = -P[1] * inv, ty = -P[2] * inv, tz = -P[3] * inv;
  float* cv = canvas + (size_t)b * IMGF;
  for (int pt = t; pt < IMGF; pt += 256) {
    int ix = pt & 15, iy = (pt >> 4) & 15, iz = pt >> 8;
    float gx = ((2.f * ix + 1.f) / 16.f - 1.f) * s + tx;
    float gy = ((2.f * iy + 1.f) / 16.f - 1.f) * s + ty;
    float gz = ((2.f * iz + 1.f) / 16.f - 1.f) * s + tz;
    cv[pt] += sample3d(patch, 5, 5, 5, gx, gy, gz);
  }
}

// fp32 (rows x cols) -> bf16 (rowsPad x colsPad) zero-padded
__global__ __launch_bounds__(256)
void convert_pad_kernel(const float* __restrict__ W, bf16_t* __restrict__ Wb,
                        int rows, int cols, int rowsPad, int colsPad) {
  int idx = blockIdx.x * blockDim.x + threadIdx.x;
  if (idx >= rowsPad * colsPad) return;
  int r = idx / colsPad, c = idx - r * colsPad;
  float v = (r < rows && c < cols) ? W[(size_t)r * cols + c] : 0.f;
  Wb[idx] = (bf16_t)v;
}

// ---------------------------------------------------------------------------
// Host orchestration
// ---------------------------------------------------------------------------
static inline size_t align256(size_t x) { return (x + 255) & ~(size_t)255; }

extern "C" void kernel_launch(void* const* d_in, const int* in_sizes, int n_in,
                              void* d_out, int out_size, void* d_ws, size_t ws_size,
                              hipStream_t stream) {
  (void)in_sizes; (void)n_in; (void)out_size; (void)ws_size;
  const float* x        = (const float*)d_in[0];
  const float* e        = (const float*)d_in[1];
  const float* enc_Wih  = (const float*)d_in[2];
  const float* enc_Whh  = (const float*)d_in[3];
  const float* enc_bih  = (const float*)d_in[4];
  const float* enc_bhh  = (const float*)d_in[5];
  const float* dec_Wih  = (const float*)d_in[6];
  const float* dec_Whh  = (const float*)d_in[7];
  const float* dec_bih  = (const float*)d_in[8];
  const float* dec_bhh  = (const float*)d_in[9];
  const float* mu_W     = (const float*)d_in[10];
  const float* mu_b     = (const float*)d_in[11];
  const float* sig_W    = (const float*)d_in[12];
  const float* sig_b    = (const float*)d_in[13];
  const float* w1_W     = (const float*)d_in[14];
  const float* w1_b     = (const float*)d_in[15];
  const float* w2_W     = (const float*)d_in[16];
  const float* w2_b     = (const float*)d_in[17];
  const float* read_W   = (const float*)d_in[18];
  const float* read_b   = (const float*)d_in[19];
  float* canvas = (float*)d_out;

  char* ws = (char*)d_ws;
  size_t off = 0;
  auto alloc = [&](size_t bytes) { void* p = ws + off; off = align256(off + bytes); return p; };

  // bf16 weights
  bf16_t* WihB  = (bf16_t*)alloc((size_t)NGATE * KENC * 2);
  bf16_t* WhhB  = (bf16_t*)alloc((size_t)NGATE * ENCD * 2);
  bf16_t* dWihB = (bf16_t*)alloc((size_t)NGATE * ZDIM * 2);
  bf16_t* dWhhB = (bf16_t*)alloc((size_t)NGATE * DECD * 2);
  bf16_t* MSB   = (bf16_t*)alloc((size_t)256 * ENCD * 2);   // [mu_W ; sig_W]
  bf16_t* W2B   = (bf16_t*)alloc((size_t)128 * DECD * 2);   // w2_W row-padded
  // bf16 activations
  bf16_t* Xcat  = (bf16_t*)alloc((size_t)NBATCH * KENC * 2);
  bf16_t* hencB = (bf16_t*)alloc((size_t)NBATCH * ENCD * 2);
  bf16_t* hdecB = (bf16_t*)alloc((size_t)NBATCH * DECD * 2);
  bf16_t* zB    = (bf16_t*)alloc((size_t)NBATCH * ZDIM * 2);
  // fp32 state + scratch
  float* henc  = (float*)alloc((size_t)NBATCH * ENCD * 4);
  float* cenc  = (float*)alloc((size_t)NBATCH * ENCD * 4);
  float* hdec  = (float*)alloc((size_t)NBATCH * DECD * 4);
  float* cdec  = (float*)alloc((size_t)NBATCH * DECD * 4);
  float* gbuf  = (float*)alloc((size_t)NBATCH * NGATE * 4);
  float* msbuf = (float*)alloc((size_t)NBATCH * 256 * 4);
  float* pbuf  = (float*)alloc((size_t)NBATCH * 128 * 4);

  // zero-init state + canvas (harness poisons buffers)
  hipMemsetAsync(henc, 0, (size_t)NBATCH * ENCD * 4, stream);
  hipMemsetAsync(cenc, 0, (size_t)NBATCH * ENCD * 4, stream);
  hipMemsetAsync(hdec, 0, (size_t)NBATCH * DECD * 4, stream);
  hipMemsetAsync(cdec, 0, (size_t)NBATCH * DECD * 4, stream);
  hipMemsetAsync(hencB, 0, (size_t)NBATCH * ENCD * 2, stream);
  hipMemsetAsync(hdecB, 0, (size_t)NBATCH * DECD * 2, stream);
  hipMemsetAsync(canvas, 0, (size_t)NBATCH * IMGF * 4, stream);

  auto cvt = [&](const float* src, bf16_t* dst, int r, int c, int rp, int cp) {
    int total = rp * cp;
    convert_pad_kernel<<<(total + 255) / 256, 256, 0, stream>>>(src, dst, r, c, rp, cp);
  };
  cvt(enc_Wih, WihB, NGATE, KRAW, NGATE, KENC);
  cvt(enc_Whh, WhhB, NGATE, ENCD, NGATE, ENCD);
  cvt(dec_Wih, dWihB, NGATE, ZDIM, NGATE, ZDIM);
  cvt(dec_Whh, dWhhB, NGATE, DECD, NGATE, DECD);
  cvt(mu_W,  MSB,              128, ENCD, 128, ENCD);
  cvt(sig_W, MSB + 128 * ENCD, 128, ENCD, 128, ENCD);
  cvt(w2_W,  W2B, 125, DECD, 128, DECD);

  const int gemmBlk = 256;
  for (int t = 0; t < T_STEPS; ++t) {
    const float* e_t = e + (size_t)t * NBATCH * ZDIM;

    // r_t + concat -> Xcat (bf16, K padded to 640)
    read_kernel<<<NBATCH, 128, 0, stream>>>(x, hdec, read_W, read_b, Xcat);

    // enc gates: Xcat @ Wih^T + h_enc @ Whh^T + bih + bhh
    wmma_gemm_kernel<<<(NBATCH / 64) * (NGATE / 128), gemmBlk, 0, stream>>>(
        Xcat, WihB, KENC, hencB, WhhB, ENCD, enc_bih, enc_bhh, gbuf, NGATE);
    lstm_pointwise_kernel<<<(NBATCH * ENCD) / 256, 256, 0, stream>>>(
        gbuf, henc, cenc, hencB, ENCD);

    // [mu ; log_sigma] = h_enc @ [mu_W ; sig_W]^T
    wmma_gemm_kernel<<<(NBATCH / 64) * (256 / 128), gemmBlk, 0, stream>>>(
        hencB, MSB, ENCD, nullptr, nullptr, 0, nullptr, nullptr, msbuf, 256);
    z_kernel<<<(NBATCH * ZDIM) / 256, 256, 0, stream>>>(msbuf, mu_b, sig_b, e_t, zB);

    // dec gates: z @ dec_Wih^T + h_dec @ dec_Whh^T + bih + bhh
    wmma_gemm_kernel<<<(NBATCH / 64) * (NGATE / 128), gemmBlk, 0, stream>>>(
        zB, dWihB, ZDIM, hdecB, dWhhB, DECD, dec_bih, dec_bhh, gbuf, NGATE);
    lstm_pointwise_kernel<<<(NBATCH * DECD) / 256, 256, 0, stream>>>(
        gbuf, hdec, cdec, hdecB, DECD);

    // write patch = h_dec @ w2_W^T (N padded to 128), then splat into canvas
    wmma_gemm_kernel<<<(NBATCH / 64) * (128 / 128), gemmBlk, 0, stream>>>(
        hdecB, W2B, DECD, nullptr, nullptr, 0, nullptr, nullptr, pbuf, 128);
    write_kernel<<<NBATCH, 256, 0, stream>>>(pbuf, w2_b, hdec, w1_W, w1_b, canvas);
  }
}
